// NeuralHMM_55319178773183
// MI455X (gfx1250) — matrix operand order repaired and checked by reading, not compiled, not run on hardware
//
#include <hip/hip_runtime.h>
#include <math.h>

// Problem constants (from reference)
#define B_   8
#define N_   128
#define E_   512
#define D_   80
#define T_   256
#define M_   512     // LSTM hidden
#define P_   256
#define G4   2048    // 4*M
#define OUTC 161     // 2*D+1
#define OUTP 176     // padded to 11 tiles of 16
#define NEGV  (-1.0e10f)
#define EPS_  (1.0e-4f)
#define STDF  (0.001f)

typedef _Float16 half8  __attribute__((ext_vector_type(8)));
typedef _Float16 half16 __attribute__((ext_vector_type(16)));
typedef float    float8 __attribute__((ext_vector_type(8)));

// ---------------- WMMA fragment helpers (CDNA5 16x16x32 f16, wave32) -------
// A: row-major (M x K) f16, lda in halves. ISA 16-bit A 16x32 layout:
//   lanes 0-15 row M=lane, lanes 16-31 row M=lane-16
//   halves 0-7 : K = k0 + (lane>=16 ? 8:0) + j
//   halves 8-15: K = k0 + 16 + (lane>=16 ? 8:0) + j
__device__ inline half16 load_a_frag(const _Float16* A, int lda, int m0, int k0, int lane) {
  int row  = m0 + (lane & 15);
  int koff = k0 + ((lane & 16) ? 8 : 0);
  const _Float16* p = A + row * lda + koff;
  half8 lo = *(const half8*)(p);
  half8 hi = *(const half8*)(p + 16);
  return __builtin_shufflevector(lo, hi, 0,1,2,3,4,5,6,7,8,9,10,11,12,13,14,15);
}
// B: weight stored (N x K) row-major f16 (i.e. Bᵀ of the math B). 32x16 layout:
//   lane holds column N = lane&15; halves 0-7: K = k0 + (lane>=16 ? 8:0) + j, halves 8-15: K = k0 + 16 + (lane>=16 ? 8:0) + j
__device__ inline half16 load_b_frag(const _Float16* Bw, int ldb, int n0, int k0, int lane) {
  int col  = n0 + (lane & 15);
  int koff = k0 + ((lane & 16) ? 8 : 0);
  const _Float16* p = Bw + col * ldb + koff;
  half8 lo = *(const half8*)(p);
  half8 hi = *(const half8*)(p + 16);
  return __builtin_shufflevector(lo, hi, 0,1,2,3,4,5,6,7,8,9,10,11,12,13,14,15);
}
__device__ inline float8 wmma16(half16 a, half16 b, float8 c) {
  return __builtin_amdgcn_wmma_f32_16x16x32_f16(false, a, false, b, (short)0, c, false, false);
}

// ---------------- small math helpers ----------------
__device__ inline float sigf(float x)      { return 1.0f / (1.0f + __expf(-x)); }
__device__ inline float softplusf(float x) { return (x > 20.0f) ? x : log1pf(__expf(x)); }
__device__ inline float logclipsig(float x){ return __logf(fmaxf(sigf(x), EPS_)); }
__device__ inline float logaddexpf_(float a, float b) {
  float m = fmaxf(a, b);
  return m + log1pf(__expf(-fabsf(a - b)));
}

// ---------------- Kernel 1: stage f16 copies of weights/activations --------
__global__ __launch_bounds__(256) void stage_kernel(
    const float* __restrict__ ow1, const float* __restrict__ ow2,
    const float* __restrict__ wih, const float* __restrict__ whh,
    const float* __restrict__ inputs, const float* __restrict__ ow0,
    _Float16* __restrict__ ow1h, _Float16* __restrict__ ow2h,
    _Float16* __restrict__ wihh, _Float16* __restrict__ whhh,
    _Float16* __restrict__ in16, _Float16* __restrict__ ow0Eh)
{
  int i = blockIdx.x * blockDim.x + threadIdx.x;
  int stride = gridDim.x * blockDim.x;
  for (int idx = i; idx < P_ * P_; idx += stride) ow1h[idx] = (_Float16)ow1[idx];
  for (int idx = i; idx < OUTP * P_; idx += stride) {
    int r = idx / P_;
    ow2h[idx] = (r < OUTC) ? (_Float16)ow2[idx] : (_Float16)0.0f;
  }
  for (int idx = i; idx < G4 * P_; idx += stride) wihh[idx] = (_Float16)wih[idx];
  for (int idx = i; idx < G4 * M_; idx += stride) whhh[idx] = (_Float16)whh[idx];
  for (int idx = i; idx < B_ * N_ * E_; idx += stride) in16[idx] = (_Float16)inputs[idx];
  for (int idx = i; idx < P_ * E_; idx += stride) {
    int j = idx / E_, k = idx % E_;
    ow0Eh[idx] = (_Float16)ow0[j * (E_ + M_) + k];   // first E columns of ow0
  }
}

// ---------------- Kernel 2: prenet for all (t,b) rows ----------------------
__global__ __launch_bounds__(256) void prenet_kernel(
    const float* __restrict__ mels, const float* __restrict__ w0,
    const float* __restrict__ w1, _Float16* __restrict__ pre2h)
{
  __shared__ float s_x[D_];
  __shared__ float s_pre[P_];
  int row = blockIdx.x;          // t*B + b
  int t = row >> 3, b = row & 7;
  int tid = threadIdx.x;
  if (tid < D_) s_x[tid] = (t == 0) ? 0.0f : mels[(b * D_ + tid) * T_ + (t - 1)];
  __syncthreads();
  float acc = 0.0f;
  for (int d = 0; d < D_; ++d) acc += s_x[d] * w0[tid * D_ + d];
  s_pre[tid] = fmaxf(acc, 0.0f);
  __syncthreads();
  float acc2 = 0.0f;
  for (int k = 0; k < P_; ++k) acc2 += s_pre[k] * w1[tid * P_ + k];
  pre2h[row * P_ + tid] = (_Float16)fmaxf(acc2, 0.0f);
}

// ---------------- Kernel 3: generic WMMA GEMM  C = A @ Bw^T + bias ---------
__global__ __launch_bounds__(256) void wmma_gemm_kernel(
    const _Float16* __restrict__ A, int lda,
    const _Float16* __restrict__ Bw, int ldb,
    const float* __restrict__ bias0, const float* __restrict__ bias1,
    float* __restrict__ C, int ldc,
    int mtiles, int ntiles, int K)
{
  int wave = threadIdx.x >> 5;
  int lane = threadIdx.x & 31;
  int tile = blockIdx.x * 8 + wave;
  if (tile >= mtiles * ntiles) return;   // wave-uniform
  int m0 = (tile / ntiles) * 16;
  int n0 = (tile % ntiles) * 16;
  float8 acc = {};
  for (int k0 = 0; k0 < K; k0 += 32) {
    half16 af = load_a_frag(A, lda, m0, k0, lane);
    half16 bf = load_b_frag(Bw, ldb, n0, k0, lane);
    acc = wmma16(af, bf, acc);
  }
  int col = n0 + (lane & 15);
  float bb = 0.0f;
  if (bias0) bb += bias0[col];
  if (bias1) bb += bias1[col];
  int rb = m0 + ((lane & 16) ? 8 : 0);
#pragma unroll
  for (int r = 0; r < 8; ++r) C[(rb + r) * ldc + col] = acc[r] + bb;
}

// ---------------- Kernel 4: persistent per-batch sequential decoder --------
// grid = B_ blocks x 512 threads (16 waves). Each block owns one batch chain.
__global__ __launch_bounds__(512) void seq_kernel(
    const float* __restrict__ mels,
    const _Float16* __restrict__ whhh,    // (2048, 512) f16
    const float* __restrict__ gates_pre,  // (T*B, 2048) f32, biases folded
    const float* __restrict__ zbase,      // (B*N, 256) f32, ob0 folded
    const _Float16* __restrict__ ow0Eh,   // (256, 512) f16 (cols E.. of ow0 NOT here;
                                          //  unused in this kernel)
    const _Float16* __restrict__ ow0Mh,   // (256, 512) f16: ow0[:, E:E+M]
    const _Float16* __restrict__ ow1h,    // (256,256) f16
    const float* __restrict__ ob1,
    const _Float16* __restrict__ ow2h,    // (176,256) f16 (rows >=161 zero)
    const float* __restrict__ ob2,
    const int* __restrict__ inputs_len,
    const int* __restrict__ mel_lens,
    float* __restrict__ out)              // [B] logprob, then (B,T,N) la
{
  __shared__ __align__(16) _Float16 s_a [N_ * P_];   // 64KB; reused as p (f16 128x176)
  __shared__ __align__(16) _Float16 s_a2[N_ * P_];   // 64KB
  __shared__ __align__(16) _Float16 s_hA[16 * M_];   // 16KB: A tile, row0 = h, rest 0
  __shared__ float s_h[M_], s_c[M_];
  __shared__ float s_gates[G4];
  __shared__ float s_hproj[P_];
  __shared__ float s_xt[D_];
  __shared__ float s_tv[N_];
  __shared__ float s_la[N_];
  __shared__ float s_lat[N_];
  __shared__ float s_logc;

  const int b    = blockIdx.x;
  const int tid  = threadIdx.x;
  const int wave = tid >> 5;
  const int lane = tid & 31;
  const int ilen = inputs_len[b];
  const int mlen = mel_lens[b];

  if (tid < M_) { s_h[tid] = 0.0f; s_c[tid] = 0.0f; }
  if (tid < N_) s_la[tid] = 0.0f;
  for (int idx = tid; idx < 16 * M_; idx += 512) s_hA[idx] = (_Float16)0.0f;
  __syncthreads();

  float logprob = 0.0f;

  for (int t = 0; t < T_; ++t) {
    const float* gp = gates_pre + (size_t)(t * B_ + b) * G4;
    // prefetch the adjacent gates_pre row (t*B+b+1, 8KB) into caches
    if (t + 1 < T_ && tid < 16)
      __builtin_prefetch((const void*)(gp + G4 + tid * 128), 0, 3);

    // ---- 1. gates = gates_pre[t,b] + h_{t-1} @ whh^T via WMMA ----
    // A tile = s_hA (row0 = h, rows 1..15 zero). 128 column tiles over 16
    // waves: wave handles tiles wave+16*i; shared A-fragment hoisted.
    {
      float8 accs[8];
#pragma unroll
      for (int i = 0; i < 8; ++i) accs[i] = (float8){};
#pragma unroll 1
      for (int k0 = 0; k0 < M_; k0 += 32) {
        half16 af = load_a_frag(s_hA, M_, 0, k0, lane);
#pragma unroll
        for (int i = 0; i < 8; ++i) {
          int n0 = (wave + 16 * i) * 16;
          half16 bf = load_b_frag(whhh, M_, n0, k0, lane);
          accs[i] = wmma16(af, bf, accs[i]);
        }
      }
      if (lane < 16) {
#pragma unroll
        for (int i = 0; i < 8; ++i) {
          int g = (wave + 16 * i) * 16 + lane;   // row M=0 lives in acc[0], lanes 0..15
          s_gates[g] = accs[i][0] + gp[g];
        }
      }
    }
    __syncthreads();
    // ---- 2. LSTM cell update (order: i, f, g, o); refresh h row of s_hA ----
    {
      float ig = s_gates[tid];
      float fg = s_gates[M_ + tid];
      float gg = s_gates[2 * M_ + tid];
      float og = s_gates[3 * M_ + tid];
      float c  = sigf(fg) * s_c[tid] + sigf(ig) * tanhf(gg);
      float h  = sigf(og) * tanhf(c);
      s_c[tid]  = c;
      s_h[tid]  = h;
      s_hA[tid] = (_Float16)h;            // row 0 of the 16xK A tile
      if (tid < D_) s_xt[tid] = mels[(b * D_ + tid) * T_ + t];
    }
    __syncthreads();
    // ---- 3. hproj = h_t @ ow0[:, E:]^T via WMMA (16 tiles, 1 per wave) ----
    {
      int n0 = wave * 16;
      float8 acc = {};
#pragma unroll
      for (int k0 = 0; k0 < M_; k0 += 32) {
        half16 af = load_a_frag(s_hA, M_, 0, k0, lane);
        half16 bf = load_b_frag(ow0Mh, M_, n0, k0, lane);
        acc = wmma16(af, bf, acc);
      }
      if (lane < 16) s_hproj[n0 + lane] = acc[0];
    }
    __syncthreads();
    // ---- 4. a = relu(zbase + hproj) -> f16 LDS ----
    for (int idx = tid; idx < N_ * P_; idx += 512) {
      int n = idx >> 8, j = idx & 255;
      float v = zbase[(size_t)(b * N_ + n) * P_ + j] + s_hproj[j];
      s_a[idx] = (_Float16)fmaxf(v, 0.0f);
    }
    __syncthreads();
    // ---- 5. GEMM1: a2 = relu(a @ ow1^T + ob1) ; 8x16 tiles over 16 waves ----
#pragma unroll 1
    for (int i = 0; i < 8; ++i) {
      int tile = wave + i * 16;
      int m0 = (tile >> 4) * 16;
      int n0 = (tile & 15) * 16;
      float8 acc = {};
#pragma unroll
      for (int k0 = 0; k0 < P_; k0 += 32) {
        half16 af = load_a_frag(s_a,  P_, m0, k0, lane);
        half16 bf = load_b_frag(ow1h, P_, n0, k0, lane);
        acc = wmma16(af, bf, acc);
      }
      int col = n0 + (lane & 15);
      float bb = ob1[col];
      int rb = m0 + ((lane & 16) ? 8 : 0);
#pragma unroll
      for (int r = 0; r < 8; ++r)
        s_a2[(rb + r) * P_ + col] = (_Float16)fmaxf(acc[r] + bb, 0.0f);
    }
    __syncthreads();
    // ---- 6. GEMM2: p = a2 @ ow2^T + ob2 ; 8x11 tiles, p stored f16 over s_a ----
    _Float16* s_p = s_a;   // safe: s_a no longer read this step
#pragma unroll 1
    for (int i = 0; i < 6; ++i) {
      int tile = wave + i * 16;
      if (tile < 88) {           // wave-uniform guard
        int m0 = (tile / 11) * 16;
        int n0 = (tile % 11) * 16;
        float8 acc = {};
#pragma unroll
        for (int k0 = 0; k0 < P_; k0 += 32) {
          half16 af = load_a_frag(s_a2, P_, m0, k0, lane);
          half16 bf = load_b_frag(ow2h, P_, n0, k0, lane);
          acc = wmma16(af, bf, acc);
        }
        int col = n0 + (lane & 15);
        float bb = (col < OUTC) ? ob2[col] : 0.0f;
        int rb = m0 + ((lane & 16) ? 8 : 0);
#pragma unroll
        for (int r = 0; r < 8; ++r)
          s_p[(rb + r) * OUTP + col] = (_Float16)(acc[r] + bb);
      }
    }
    __syncthreads();
    // ---- 7a. emission per state ----
    if (tid < N_) {
      int n = tid;
      float em = 0.0f;
      if (n < ilen) {
        const float HALF_LOG2PI = 0.9189385332046727f;
        float s = 0.0f;
        for (int d = 0; d < D_; ++d) {
          float mean = (float)s_p[n * OUTP + d];
          float sh   = (float)s_p[n * OUTP + D_ + d];
          float std  = softplusf(sh) + STDF;
          float z    = (s_xt[d] - mean) / std;
          s += -0.5f * z * z - __logf(std);
        }
        em = s - HALF_LOG2PI * (float)D_;
      }
      s_lat[n] = em;                      // stash em
      s_tv[n]  = (float)s_p[n * OUTP + 2 * D_];
    }
    __syncthreads();
    // ---- 7b. HMM forward update ----
    if (tid < N_) {
      int n = tid;
      float em = s_lat[n];
      float la;
      if (t == 0) {
        la = ((n == 0) ? 0.0f : NEGV) + em;
      } else {
        float stay  = s_la[n] + logclipsig(-s_tv[n]);
        float leave = (n == 0) ? NEGV : (s_la[n - 1] + logclipsig(s_tv[n - 1]));
        float o = logaddexpf_(stay, leave);
        o = (n < ilen) ? o : NEGV;
        la = em + o;
      }
      s_lat[n] = la;
    }
    __syncthreads();
    // ---- 7c. logsumexp over states (serial; negligible vs GEMMs) ----
    if (tid == 0) {
      float m = s_lat[0];
      for (int n = 1; n < N_; ++n) m = fmaxf(m, s_lat[n]);
      float s = 0.0f;
      for (int n = 0; n < N_; ++n) s += __expf(s_lat[n] - m);
      s_logc = m + __logf(s);
    }
    __syncthreads();
    // ---- 7d. normalize, emit outputs ----
    if (tid < N_) {
      float las = s_lat[tid] - s_logc;
      s_la[tid] = las;
      out[B_ + (size_t)(b * T_ + t) * N_ + tid] = las;
    }
    if (tid == 0 && t < mlen) logprob += s_logc;
    __syncthreads();
  }
  if (tid == 0) out[b] = logprob;
}

// ---------------- Kernel 5: stage ow0[:, E:] to f16 ------------------------
__global__ __launch_bounds__(256) void stage_ow0M_kernel(
    const float* __restrict__ ow0, _Float16* __restrict__ ow0Mh)
{
  int i = blockIdx.x * blockDim.x + threadIdx.x;
  int stride = gridDim.x * blockDim.x;
  for (int idx = i; idx < P_ * M_; idx += stride) {
    int j = idx / M_, k = idx % M_;
    ow0Mh[idx] = (_Float16)ow0[j * (E_ + M_) + E_ + k];
  }
}

// ---------------- host launcher --------------------------------------------
extern "C" void kernel_launch(void* const* d_in, const int* in_sizes, int n_in,
                              void* d_out, int out_size, void* d_ws, size_t ws_size,
                              hipStream_t stream) {
  const float* inputs = (const float*)d_in[0];
  const float* mels   = (const float*)d_in[1];
  const float* w0     = (const float*)d_in[2];
  const float* w1     = (const float*)d_in[3];
  const float* wih    = (const float*)d_in[4];
  const float* whh    = (const float*)d_in[5];
  const float* bih    = (const float*)d_in[6];
  const float* bhh    = (const float*)d_in[7];
  const float* ow0    = (const float*)d_in[8];
  const float* ob0    = (const float*)d_in[9];
  const float* ow1    = (const float*)d_in[10];
  const float* ob1    = (const float*)d_in[11];
  const float* ow2    = (const float*)d_in[12];
  const float* ob2    = (const float*)d_in[13];
  const int*   ilen   = (const int*)d_in[14];
  const int*   mlen   = (const int*)d_in[15];
  float* out = (float*)d_out;

  char* ws = (char*)d_ws;
  size_t off = 0;
  auto carve = [&](size_t bytes) -> void* {
    void* p = (void*)(ws + off);
    off = (off + bytes + 255) & ~(size_t)255;
    return p;
  };
  _Float16* ow1h      = (_Float16*)carve((size_t)P_ * P_ * 2);
  _Float16* ow2h      = (_Float16*)carve((size_t)OUTP * P_ * 2);
  _Float16* wihh      = (_Float16*)carve((size_t)G4 * P_ * 2);
  _Float16* whhh      = (_Float16*)carve((size_t)G4 * M_ * 2);
  _Float16* in16      = (_Float16*)carve((size_t)B_ * N_ * E_ * 2);
  _Float16* ow0Eh     = (_Float16*)carve((size_t)P_ * E_ * 2);
  _Float16* ow0Mh     = (_Float16*)carve((size_t)P_ * M_ * 2);
  _Float16* pre2h     = (_Float16*)carve((size_t)T_ * B_ * P_ * 2);
  float*    gates_pre = (float*)carve((size_t)T_ * B_ * G4 * 4);
  float*    zbase     = (float*)carve((size_t)B_ * N_ * P_ * 4);
  (void)ws_size; (void)in_sizes; (void)n_in; (void)out_size;

  // 1) stage f16 copies
  stage_kernel<<<dim3(1024), dim3(256), 0, stream>>>(
      ow1, ow2, wih, whh, inputs, ow0, ow1h, ow2h, wihh, whhh, in16, ow0Eh);
  stage_ow0M_kernel<<<dim3(512), dim3(256), 0, stream>>>(ow0, ow0Mh);

  // 2) prenet for all (t,b)
  prenet_kernel<<<dim3(T_ * B_), dim3(256), 0, stream>>>(mels, w0, w1, pre2h);

  // 3) gates_pre = pre2 @ wih^T + (b_ih + b_hh)   (2048x2048, K=256)
  wmma_gemm_kernel<<<dim3((128 * 128) / 8), dim3(256), 0, stream>>>(
      pre2h, P_, wihh, P_, bih, bhh, gates_pre, G4, 128, 128, P_);

  // 4) zbase = inputs @ ow0[:, :E]^T + ob0        (1024x256, K=512)
  wmma_gemm_kernel<<<dim3((64 * 16) / 8), dim3(256), 0, stream>>>(
      in16, E_, ow0Eh, E_, ob0, (const float*)nullptr, zbase, P_, 64, 16, E_);

  // 5) per-batch persistent sequential decode (8 independent chains)
  seq_kernel<<<dim3(B_), dim3(512), 0, stream>>>(
      mels, whhh, gates_pre, zbase, ow0Eh, ow0Mh, ow1h, ob1, ow2h, ob2,
      ilen, mlen, out);
}
